// DoubleNet_12584254177926
// MI455X (gfx1250) — matrix operand order repaired and checked by reading, not compiled, not run on hardware
//
#include <hip/hip_runtime.h>

typedef __attribute__((ext_vector_type(16))) _Float16 v16h;
typedef __attribute__((ext_vector_type(8)))  float    v8f;

#define TILE_ROWS 32
#define NTHREADS  256
#define NWAVES    8

__device__ __forceinline__ float fast_tanh(float x) {
    float e = __expf(2.0f * x);
    return 1.0f - 2.0f / (e + 1.0f);
}

// Wave-local "barrier": all 8 Sinkhorn workers of a sample live in the same
// wave32, and LDS ops from one wave execute in order at the LDS unit, so
// cross-lane visibility needs only a compiler scheduling/memory fence —
// no s_barrier_signal/s_barrier_wait.
__device__ __forceinline__ void wave_sync() {
    __builtin_amdgcn_fence(__ATOMIC_ACQ_REL, "wavefront");
    __builtin_amdgcn_wave_barrier();
}

// Stage a (din x dout) fp32 weight matrix from global into LDS as f16,
// transposed to (dout x din) so B-fragment reads are contiguous in K.
__device__ __forceinline__ void stage_weights(const float* __restrict__ W,
                                              _Float16* __restrict__ Wt,
                                              int din, int doutShift, int tid) {
    const int dout  = 1 << doutShift;
    const int total = din << doutShift;
    for (int idx = tid; idx < total; idx += NTHREADS) {
        const int k = idx >> doutShift;      // coalesced global read W[k*dout+n]
        const int n = idx & (dout - 1);
        Wt[n * din + k] = (_Float16)W[idx];
    }
}

// One wave computes 16x16 C tiles of  act(16xK) @ Wt^T(Kxdout) + bias.
// MODE 0: tanh -> f16 LDS out.  MODE 1: raw f32 LDS out.
template <int MODE>
__device__ __forceinline__ void wave_gemm(const _Float16* __restrict__ aIn, int K,
                                          const _Float16* __restrict__ Wt,
                                          const float* __restrict__ bias,
                                          _Float16* __restrict__ hOut,
                                          float* __restrict__ fOut,
                                          int dout, int wave, int lane) {
    const int nColTiles = dout >> 4;
    const int nTiles    = (TILE_ROWS / 16) * nColTiles;
    const int half  = lane >> 4;   // lane group 0 / 1
    const int l15   = lane & 15;
    const int aKoff = half * 16;   // A 16x32 f16: contiguous K halves
    const int bKoff = half * 16;   // B 32x16 f16: contiguous K halves

    for (int tile = wave; tile < nTiles; tile += NWAVES) {
        const int mt = tile / nColTiles;
        const int nt = tile - mt * nColTiles;
        v8f c = {0.f, 0.f, 0.f, 0.f, 0.f, 0.f, 0.f, 0.f};
        const _Float16* aRow = aIn + (mt * 16 + l15) * K;  // A: lane = row M
        const _Float16* bRow = Wt  + (nt * 16 + l15) * K;  // B: lane = col N
        for (int kb = 0; kb < K; kb += 32) {
            v16h a, b;
#pragma unroll
            for (int v = 0; v < 8; ++v) {
                const int ak = kb + aKoff + 2 * v;
                a[2 * v]     = aRow[ak];
                a[2 * v + 1] = aRow[ak + 1];
                const int bk = kb + bKoff + 2 * v;
                b[2 * v]     = bRow[bk];
                b[2 * v + 1] = bRow[bk + 1];
            }
            c = __builtin_amdgcn_wmma_f32_16x16x32_f16(
                    /*neg_a=*/false, a, /*neg_b=*/false, b,
                    /*c_mod=*/(short)0, c, /*reuse_a=*/false, /*reuse_b=*/false);
        }
        const int   n  = nt * 16 + l15;
        const float bn = bias[n];
#pragma unroll
        for (int r = 0; r < 8; ++r) {                 // C layout: M = r + 8*half
            const int   m   = mt * 16 + r + half * 8;
            const float val = c[r] + bn;
            if (MODE == 0) hOut[m * dout + n] = (_Float16)fast_tanh(val);
            else           fOut[m * dout + n] = val;
        }
    }
}

__global__ __launch_bounds__(NTHREADS)
void doublenet_fused(const float* __restrict__ bids,
                     const float* __restrict__ aw0, const float* __restrict__ ab0,
                     const float* __restrict__ aw1, const float* __restrict__ ab1,
                     const float* __restrict__ aw2, const float* __restrict__ ab2,
                     const float* __restrict__ aw3, const float* __restrict__ ab3,
                     const float* __restrict__ pw0, const float* __restrict__ pb0,
                     const float* __restrict__ pw1, const float* __restrict__ pb1,
                     const float* __restrict__ pw2, const float* __restrict__ pb2,
                     const float* __restrict__ pw3, const float* __restrict__ pb3,
                     float* __restrict__ allocs_out, float* __restrict__ pay_out,
                     int bsz) {
    __shared__ _Float16 Xs[TILE_ROWS * 256];
    __shared__ _Float16 Wt[256 * 128];
    __shared__ _Float16 hA[TILE_ROWS * 128];
    __shared__ _Float16 hB[TILE_ROWS * 128];
    __shared__ float    Mbuf[TILE_ROWS * 256];
    __shared__ float    p3s[TILE_ROWS * 16];
    __shared__ float    uu[TILE_ROWS * 20];
    __shared__ float    vv[TILE_ROWS * 20];

    const int tid     = threadIdx.x;
    const int wave    = tid >> 5;
    const int lane    = tid & 31;
    const int rowbase = blockIdx.x * TILE_ROWS;

    // Stage X tile (fp32 -> f16), zero-fill OOB rows.
    for (int idx = tid; idx < TILE_ROWS * 256; idx += NTHREADS) {
        const int grow = rowbase + (idx >> 8);
        Xs[idx] = (grow < bsz) ? (_Float16)bids[(size_t)rowbase * 256 + idx]
                               : (_Float16)0.f;
    }
    __syncthreads();

    // Allocation MLP: 256->128->128->128->256, tanh except last.
    stage_weights(aw0, Wt, 256, 7, tid); __syncthreads();
    wave_gemm<0>(Xs, 256, Wt, ab0, hA, nullptr, 128, wave, lane); __syncthreads();
    stage_weights(aw1, Wt, 128, 7, tid); __syncthreads();
    wave_gemm<0>(hA, 128, Wt, ab1, hB, nullptr, 128, wave, lane); __syncthreads();
    stage_weights(aw2, Wt, 128, 7, tid); __syncthreads();
    wave_gemm<0>(hB, 128, Wt, ab2, hA, nullptr, 128, wave, lane); __syncthreads();
    stage_weights(aw3, Wt, 128, 8, tid); __syncthreads();
    wave_gemm<1>(hA, 128, Wt, ab3, nullptr, Mbuf, 256, wave, lane); __syncthreads();

    // Payment MLP: 256->128->128->128->16, tanh except last (sigmoid later).
    stage_weights(pw0, Wt, 256, 7, tid); __syncthreads();
    wave_gemm<0>(Xs, 256, Wt, pb0, hA, nullptr, 128, wave, lane); __syncthreads();
    stage_weights(pw1, Wt, 128, 7, tid); __syncthreads();
    wave_gemm<0>(hA, 128, Wt, pb1, hB, nullptr, 128, wave, lane); __syncthreads();
    stage_weights(pw2, Wt, 128, 7, tid); __syncthreads();
    wave_gemm<0>(hB, 128, Wt, pb2, hA, nullptr, 128, wave, lane); __syncthreads();
    stage_weights(pw3, Wt, 128, 4, tid); __syncthreads();
    wave_gemm<1>(hA, 128, Wt, pb3, nullptr, p3s, 16, wave, lane); __syncthreads();

    // Sinkhorn in dual form: K = K0 + u_i + v_j with K0 = 10*M (0 on pad row/col).
    // Row update:  u_i = log_a_i - lse_j(K0_ij + v_j)
    // Col update:  v_j = log_b_j - lse_i(K0_ij + u_i)
    // All 8 workers of a sample live in one wave -> wave_sync() suffices.
    const int s = tid >> 3;   // sample in tile (0..31)
    const int g = tid & 7;    // worker within sample
    float* Ms = Mbuf + s * 256;
    float* us = uu + s * 20;
    float* vs = vv + s * 20;
    for (int j = g; j < 17; j += 8) vs[j] = 0.f;
    wave_sync();

    const float LOG16 = 2.7725887222397811f;
#pragma unroll 1
    for (int r = 0; r < 40; ++r) {
        for (int i = g; i < 17; i += 8) {
            float t[17];
            if (i < 16) {
                const float* Mi = Ms + i * 16;
#pragma unroll
                for (int j = 0; j < 16; ++j) t[j] = fmaf(10.f, Mi[j], vs[j]);
                t[16] = vs[16];
            } else {
#pragma unroll
                for (int j = 0; j < 17; ++j) t[j] = vs[j];
            }
            float mx = t[0];
#pragma unroll
            for (int j = 1; j < 17; ++j) mx = fmaxf(mx, t[j]);
            float sum = 0.f;
#pragma unroll
            for (int j = 0; j < 17; ++j) sum += __expf(t[j] - mx);
            us[i] = ((i < 16) ? 0.f : LOG16) - (mx + __logf(sum));
        }
        wave_sync();
        for (int j = g; j < 17; j += 8) {
            float t[17];
            if (j < 16) {
                const float* Mj = Ms + j;
#pragma unroll
                for (int i = 0; i < 16; ++i) t[i] = fmaf(10.f, Mj[i * 16], us[i]);
                t[16] = us[16];
            } else {
#pragma unroll
                for (int i = 0; i < 17; ++i) t[i] = us[i];
            }
            float mx = t[0];
#pragma unroll
            for (int i = 1; i < 17; ++i) mx = fmaxf(mx, t[i]);
            float sum = 0.f;
#pragma unroll
            for (int i = 0; i < 17; ++i) sum += __expf(t[i] - mx);
            vs[j] = ((j < 16) ? 0.f : LOG16) - (mx + __logf(sum));
        }
        wave_sync();
    }

    // allocs = exp(K0 + u + v) on the 16x16 block; payments = sigmoid(p3) * <allocs, bids>.
    const int grow = rowbase + s;
    if (grow < bsz) {
        for (int i = g; i < 16; i += 8) {
            const float* Mi   = Ms + i * 16;
            const float  ui   = us[i];
            float*       orow = allocs_out + ((size_t)grow * 16 + i) * 16;
            const float* brow = bids + (size_t)grow * 256 + i * 16;
            float dot = 0.f;
#pragma unroll
            for (int j = 0; j < 16; ++j) {
                const float aij = __expf(fmaf(10.f, Mi[j], ui + vs[j]));
                orow[j] = aij;
                dot += aij * brow[j];
            }
            const float frac = 1.f / (1.f + __expf(-p3s[s * 16 + i]));
            pay_out[(size_t)grow * 16 + i] = frac * dot;
        }
    }
}

extern "C" void kernel_launch(void* const* d_in, const int* in_sizes, int n_in,
                              void* d_out, int out_size, void* d_ws, size_t ws_size,
                              hipStream_t stream) {
    const float* bids = (const float*)d_in[0];
    const float* aw0 = (const float*)d_in[1];  const float* ab0 = (const float*)d_in[2];
    const float* aw1 = (const float*)d_in[3];  const float* ab1 = (const float*)d_in[4];
    const float* aw2 = (const float*)d_in[5];  const float* ab2 = (const float*)d_in[6];
    const float* aw3 = (const float*)d_in[7];  const float* ab3 = (const float*)d_in[8];
    const float* pw0 = (const float*)d_in[9];  const float* pb0 = (const float*)d_in[10];
    const float* pw1 = (const float*)d_in[11]; const float* pb1 = (const float*)d_in[12];
    const float* pw2 = (const float*)d_in[13]; const float* pb2 = (const float*)d_in[14];
    const float* pw3 = (const float*)d_in[15]; const float* pb3 = (const float*)d_in[16];

    const int bsz = in_sizes[0] / 256;
    float* allocs = (float*)d_out;
    float* pays   = allocs + (size_t)bsz * 256;

    const int nblocks = (bsz + TILE_ROWS - 1) / TILE_ROWS;
    hipLaunchKernelGGL(doublenet_fused, dim3(nblocks), dim3(NTHREADS), 0, stream,
                       bids, aw0, ab0, aw1, ab1, aw2, ab2, aw3, ab3,
                       pw0, pb0, pw1, pb1, pw2, pb2, pw3, pb3,
                       allocs, pays, bsz);
}
